// mLSTMblock_32676111188020
// MI455X (gfx1250) — matrix operand order repaired and checked by reading, not compiled, not run on hardware
//
#include <hip/hip_runtime.h>
#include <hip/hip_bf16.h>

// ---------------- problem constants ----------------
#define B_    2
#define S_    1024
#define E_    512
#define H_    1024
#define D_    4
#define KW    51
#define BLK_  256
#define M_    (B_ * S_)      // 2048 rows
#define EPSL  1e-5f

typedef __attribute__((ext_vector_type(16))) __bf16 v16bf;
typedef __attribute__((ext_vector_type(8)))  float  v8f;

// ---------------- helpers ----------------
__device__ __forceinline__ unsigned short f2bf(float f) {
  unsigned int u = __float_as_uint(f);
  u += 0x7FFFu + ((u >> 16) & 1u);        // round-to-nearest-even
  return (unsigned short)(u >> 16);
}

__device__ __forceinline__ float block_sum(float v, float* sbuf) {
  int t = threadIdx.x;
  sbuf[t] = v; __syncthreads();
  for (int s = 128; s > 0; s >>= 1) {
    if (t < s) sbuf[t] += sbuf[t + s];
    __syncthreads();
  }
  float r = sbuf[0]; __syncthreads();
  return r;
}

__device__ __forceinline__ float block_maxr(float v, float* sbuf) {
  int t = threadIdx.x;
  sbuf[t] = v; __syncthreads();
  for (int s = 128; s > 0; s >>= 1) {
    if (t < s) sbuf[t] = fmaxf(sbuf[t], sbuf[t + s]);
    __syncthreads();
  }
  float r = sbuf[0]; __syncthreads();
  return r;
}

// Fragment loaders per CDNA5 ISA 7.12.2 (wave32):
//  A 16x32 bf16: half h=lane>>4, r=lane&15, row M=r,
//   elements e=0..7 -> k = h*8+e, e=8..15 -> k = 16+h*8+(e-8)
//  B 32x16 bf16: col N=lane&15, elements e=0..7 -> k = h*8+e, e=8..15 -> k = 16+h*8+(e-8)
__device__ __forceinline__ v16bf load_afrag(const unsigned short* rowbase, int half) {
  v16bf a;
  uint4* p = (uint4*)&a;
  p[0] = *(const uint4*)(rowbase + half * 8);
  p[1] = *(const uint4*)(rowbase + 16 + half * 8);
  return a;
}
__device__ __forceinline__ v16bf load_bfrag(const unsigned short* rowbase, int half) {
  v16bf b;
  uint4* p = (uint4*)&b;
  p[0] = *(const uint4*)(rowbase + half * 8);
  p[1] = *(const uint4*)(rowbase + 16 + half * 8);
  return b;
}

// One K=32 step for a wave's 32x64 tile: 2 A-frags, 4 B-frags, 8 WMMAs.
__device__ __forceinline__ void wmma_tile8(const unsigned short* __restrict__ ldsA,
                                           const unsigned short* __restrict__ ldsB,
                                           int wr, int wc, int r, int half,
                                           v8f acc[2][4])
{
  v16bf af[2], bfr[4];
  for (int mi = 0; mi < 2; ++mi)
    af[mi] = load_afrag(&ldsA[(wr * 32 + mi * 16 + r) * 32], half);
  for (int ni = 0; ni < 4; ++ni)
    bfr[ni] = load_bfrag(&ldsB[(wc * 64 + ni * 16 + r) * 32], half);
  for (int mi = 0; mi < 2; ++mi)
    for (int ni = 0; ni < 4; ++ni)
      acc[mi][ni] = __builtin_amdgcn_wmma_f32_16x16x32_bf16(
          false, af[mi], false, bfr[ni], (short)0, acc[mi][ni], false, false);
}

// ---------------- generic bf16 WMMA GEMM (double-buffered) ----------------
// C[m,n] = act( sum_k A[m,k] * W[n,k] + bias[n] ), act: 0=none, 1=silu
// Block tile 128x128, K-step 32; 8 waves, each 32x64. One barrier per k-step.
__global__ __launch_bounds__(256)
void gemm_bf16_wmma(const unsigned short* __restrict__ A, int lda,
                    const unsigned short* __restrict__ W, int ldb,
                    const float* __restrict__ bias,
                    float* __restrict__ Cf, unsigned short* __restrict__ Cb,
                    int ldc, int Kd, int act)
{
  __shared__ unsigned short ldsA[2][128 * 32];
  __shared__ unsigned short ldsB[2][128 * 32];

  const int t    = threadIdx.x;
  const int lane = t & 31;
  const int wave = t >> 5;
  const int wr   = wave >> 1;          // 0..3
  const int wc   = wave & 1;           // 0..1
  const int half = lane >> 4;
  const int r    = lane & 15;
  const int mBase = blockIdx.y * 128;
  const int nBase = blockIdx.x * 128;

  // each thread stages two 16B chunks per operand tile; hoisted pointers,
  // bumped by 32 elements (64B) per k-step -> no u64 muls in the loop
  const int r0 = t >> 2,         o0 = (t & 3) << 3;
  const int r1 = (t + 256) >> 2, o1 = ((t + 256) & 3) << 3;
  const unsigned short* pA0 = A + (size_t)(mBase + r0) * lda + o0;
  const unsigned short* pA1 = A + (size_t)(mBase + r1) * lda + o1;
  const unsigned short* pB0 = W + (size_t)(nBase + r0) * ldb + o0;
  const unsigned short* pB1 = W + (size_t)(nBase + r1) * ldb + o1;
  const int sA0 = r0 * 32 + o0, sA1 = r1 * 32 + o1;

  v8f acc[2][4];
  for (int mi = 0; mi < 2; ++mi)
    for (int ni = 0; ni < 4; ++ni)
      for (int j = 0; j < 8; ++j) acc[mi][ni][j] = 0.f;

  // prologue: stage k-block 0 into buffer 0
  {
    uint4 a0 = *(const uint4*)pA0; pA0 += 32;
    uint4 a1 = *(const uint4*)pA1; pA1 += 32;
    uint4 b0 = *(const uint4*)pB0; pB0 += 32;
    uint4 b1 = *(const uint4*)pB1; pB1 += 32;
    *(uint4*)&ldsA[0][sA0] = a0;
    *(uint4*)&ldsA[0][sA1] = a1;
    *(uint4*)&ldsB[0][sA0] = b0;
    *(uint4*)&ldsB[0][sA1] = b1;
  }
  __syncthreads();

  int cur = 0;
  for (int kb = 32; kb < Kd; kb += 32) {
    // prefetch next tile into registers (overlaps with WMMA below)
    uint4 a0 = *(const uint4*)pA0; pA0 += 32;
    uint4 a1 = *(const uint4*)pA1; pA1 += 32;
    uint4 b0 = *(const uint4*)pB0; pB0 += 32;
    uint4 b1 = *(const uint4*)pB1; pB1 += 32;

    wmma_tile8(ldsA[cur], ldsB[cur], wr, wc, r, half, acc);

    *(uint4*)&ldsA[cur ^ 1][sA0] = a0;
    *(uint4*)&ldsA[cur ^ 1][sA1] = a1;
    *(uint4*)&ldsB[cur ^ 1][sA0] = b0;
    *(uint4*)&ldsB[cur ^ 1][sA1] = b1;
    __syncthreads();
    cur ^= 1;
  }
  wmma_tile8(ldsA[cur], ldsB[cur], wr, wc, r, half, acc);

  for (int mi = 0; mi < 2; ++mi)
    for (int ni = 0; ni < 4; ++ni) {
      int col = nBase + wc * 64 + ni * 16 + r;
      float bv = bias ? bias[col] : 0.f;
      for (int j = 0; j < 8; ++j) {
        int rowg = mBase + wr * 32 + mi * 16 + half * 8 + j;
        float v = acc[mi][ni][j] + bv;
        if (act == 1) v = v / (1.f + __expf(-v));     // silu
        size_t idx = (size_t)rowg * ldc + col;
        if (Cf) Cf[idx] = v;
        if (Cb) Cb[idx] = f2bf(v);
      }
    }
}

// ---------------- causal conv1d as 51 accumulated GEMMs ----------------
// left_left[t,o] = silu( sum_tap sum_i left[t-50+tap, i] * Wt[tap][o][i] + b[o] )
// Wt pre-transposed to [tap][out][in] bf16; double-buffered over the
// flattened (tap, k-step) iteration space (51*32 = 1632 steps).
__device__ __forceinline__ uint4 conv_loadA(const unsigned short* __restrict__ A,
                                            int bat, int pos, int tap, int kboff) {
  int p = pos - (KW - 1) + tap;
  if (p < 0) return make_uint4(0u, 0u, 0u, 0u);
  return *(const uint4*)&A[(size_t)((unsigned)(((bat << 10) + p) << 10) + kboff)];
}

__global__ __launch_bounds__(256)
void conv_causal_wmma(const unsigned short* __restrict__ A,   // left bf16 [M_][H_]
                      const unsigned short* __restrict__ Wt,  // [KW][H_][H_]
                      const float* __restrict__ bias,
                      float* __restrict__ Cf, unsigned short* __restrict__ Cb)
{
  __shared__ unsigned short ldsA[2][128 * 32];
  __shared__ unsigned short ldsB[2][128 * 32];

  const int t    = threadIdx.x;
  const int lane = t & 31;
  const int wave = t >> 5;
  const int wr   = wave >> 1;
  const int wc   = wave & 1;
  const int half = lane >> 4;
  const int r    = lane & 15;
  const int mBase = blockIdx.y * 128;
  const int nBase = blockIdx.x * 128;

  const int r0 = t >> 2,         o0 = (t & 3) << 3;
  const int r1 = (t + 256) >> 2, o1 = ((t + 256) & 3) << 3;
  const int bat0 = (mBase + r0) >> 10, pos0 = (mBase + r0) & (S_ - 1);
  const int bat1 = (mBase + r1) >> 10, pos1 = (mBase + r1) & (S_ - 1);
  const int bB0 = (nBase + r0) << 10;     // row * H_
  const int bB1 = (nBase + r1) << 10;
  const int sA0 = r0 * 32 + o0, sA1 = r1 * 32 + o1;

  v8f acc[2][4];
  for (int mi = 0; mi < 2; ++mi)
    for (int ni = 0; ni < 4; ++ni)
      for (int j = 0; j < 8; ++j) acc[mi][ni][j] = 0.f;

  const int NIT = KW * (H_ / 32);         // 1632 k-steps total

  // prologue: iteration 0 (tap=0, kb=0)
  {
    uint4 a0 = conv_loadA(A, bat0, pos0, 0, o0);
    uint4 a1 = conv_loadA(A, bat1, pos1, 0, o1);
    uint4 b0 = *(const uint4*)&Wt[(size_t)(bB0 + o0)];
    uint4 b1 = *(const uint4*)&Wt[(size_t)(bB1 + o1)];
    *(uint4*)&ldsA[0][sA0] = a0;
    *(uint4*)&ldsA[0][sA1] = a1;
    *(uint4*)&ldsB[0][sA0] = b0;
    *(uint4*)&ldsB[0][sA1] = b1;
  }
  __syncthreads();

  int cur = 0;
  for (int it = 1; it < NIT; ++it) {
    const int tap = it >> 5;              // 32 k-steps per tap
    const int kb  = (it & 31) << 5;
    uint4 a0 = conv_loadA(A, bat0, pos0, tap, kb + o0);
    uint4 a1 = conv_loadA(A, bat1, pos1, tap, kb + o1);
    uint4 b0 = *(const uint4*)&Wt[((size_t)tap << 20) + (bB0 + kb + o0)];
    uint4 b1 = *(const uint4*)&Wt[((size_t)tap << 20) + (bB1 + kb + o1)];

    wmma_tile8(ldsA[cur], ldsB[cur], wr, wc, r, half, acc);

    *(uint4*)&ldsA[cur ^ 1][sA0] = a0;
    *(uint4*)&ldsA[cur ^ 1][sA1] = a1;
    *(uint4*)&ldsB[cur ^ 1][sA0] = b0;
    *(uint4*)&ldsB[cur ^ 1][sA1] = b1;
    __syncthreads();
    cur ^= 1;
  }
  wmma_tile8(ldsA[cur], ldsB[cur], wr, wc, r, half, acc);

  for (int mi = 0; mi < 2; ++mi)
    for (int ni = 0; ni < 4; ++ni) {
      int col = nBase + wc * 64 + ni * 16 + r;
      float bv = bias[col];
      for (int j = 0; j < 8; ++j) {
        int rowg = mBase + wr * 32 + mi * 16 + half * 8 + j;
        float v = acc[mi][ni][j] + bv;
        v = v / (1.f + __expf(-v));                   // silu
        size_t idx = (size_t)rowg * H_ + col;
        Cf[idx] = v;
        Cb[idx] = f2bf(v);
      }
    }
}

// ---------------- pointwise / LN kernels ----------------
__global__ __launch_bounds__(256)
void cast_f32_bf16(const float* __restrict__ in, unsigned short* __restrict__ out, size_t n) {
  for (size_t i = (size_t)blockIdx.x * 256 + threadIdx.x; i < n; i += (size_t)gridDim.x * 256)
    out[i] = f2bf(in[i]);
}

// conv_w [O][I][KW] f32  ->  [KW][O][I] bf16
__global__ __launch_bounds__(256)
void conv_transpose_cast(const float* __restrict__ in, unsigned short* __restrict__ out) {
  size_t total = (size_t)KW * H_ * H_;
  for (size_t idx = (size_t)blockIdx.x * 256 + threadIdx.x; idx < total;
       idx += (size_t)gridDim.x * 256) {
    int i    = (int)(idx % H_);
    size_t q = idx / H_;
    int o    = (int)(q % H_);
    int tap  = (int)(q / H_);
    out[idx] = f2bf(in[((size_t)o * H_ + i) * KW + tap]);
  }
}

__global__ __launch_bounds__(256)
void zero_f32(float* __restrict__ p, int n) {
  for (int i = blockIdx.x * 256 + threadIdx.x; i < n; i += gridDim.x * 256) p[i] = 0.f;
}

// block per row: LN over C (C = 512 or 1024, divisible by 256)
__global__ __launch_bounds__(256)
void ln_rows(const float* __restrict__ in, float* __restrict__ outf,
             unsigned short* __restrict__ outb, int C)
{
  __shared__ float sbuf[256];
  int row = blockIdx.x, t = threadIdx.x;
  int nl = C >> 8;
  float vals[4];
  float s = 0.f, s2 = 0.f;
  for (int i = 0; i < nl; ++i) {
    float v = in[(size_t)row * C + t + (i << 8)];
    vals[i] = v; s += v; s2 += v * v;
  }
  s  = block_sum(s,  sbuf);
  s2 = block_sum(s2, sbuf);
  float mean = s / C;
  float rstd = rsqrtf(s2 / C - mean * mean + EPSL);
  for (int i = 0; i < nl; ++i) {
    float o = (vals[i] - mean) * rstd;
    size_t idx = (size_t)row * C + t + (i << 8);
    if (outf) outf[idx] = o;
    if (outb) outb[idx] = f2bf(o);
  }
}

// in-place LN then activation (1=exp, 2=sigmoid); C = H_
__global__ __launch_bounds__(256)
void ln_act(float* __restrict__ data, int mode)
{
  __shared__ float sbuf[256];
  int row = blockIdx.x, t = threadIdx.x;
  float vals[4];
  float s = 0.f, s2 = 0.f;
  for (int i = 0; i < 4; ++i) {
    float v = data[(size_t)row * H_ + t + (i << 8)];
    vals[i] = v; s += v; s2 += v * v;
  }
  s  = block_sum(s,  sbuf);
  s2 = block_sum(s2, sbuf);
  float mean = s / H_;
  float rstd = rsqrtf(s2 / H_ - mean * mean + EPSL);
  for (int i = 0; i < 4; ++i) {
    float o = (vals[i] - mean) * rstd;
    o = (mode == 1) ? __expf(o) : 1.f / (1.f + __expf(-o));
    data[(size_t)row * H_ + t + (i << 8)] = o;
  }
}

// per-row: ct = LN(i*v*k), nt = LN(i*k); accumulate column means via atomics
__global__ __launch_bounds__(256)
void ctnt_accum(const float* __restrict__ ig, const float* __restrict__ vv,
                const float* __restrict__ kk,
                float* __restrict__ ctacc, float* __restrict__ ntacc)
{
  __shared__ float sbuf[256];
  int row = blockIdx.x, t = threadIdx.x;
  float cv[4], nv[4];
  float cs = 0.f, cs2 = 0.f, ns = 0.f, ns2 = 0.f;
  for (int i = 0; i < 4; ++i) {
    size_t idx = (size_t)row * H_ + t + (i << 8);
    float a = ig[idx], b = vv[idx], c = kk[idx];
    float ct = a * b * c, nt = a * c;
    cv[i] = ct; nv[i] = nt;
    cs += ct; cs2 += ct * ct;
    ns += nt; ns2 += nt * nt;
  }
  cs  = block_sum(cs,  sbuf);
  cs2 = block_sum(cs2, sbuf);
  ns  = block_sum(ns,  sbuf);
  ns2 = block_sum(ns2, sbuf);
  float cm = cs / H_, crs = rsqrtf(cs2 / H_ - cm * cm + EPSL);
  float nm = ns / H_, nrs = rsqrtf(ns2 / H_ - nm * nm + EPSL);
  const float invR = 1.f / (float)M_;
  for (int i = 0; i < 4; ++i) {
    int col = t + (i << 8);
    atomicAdd(&ctacc[col], (cv[i] - cm) * crs * invR);
    atomicAdd(&ntacc[col], (nv[i] - nm) * nrs * invR);
  }
}

__global__ __launch_bounds__(256)
void max_partial(const float* __restrict__ q, const float* __restrict__ ntv,
                 float* __restrict__ part)
{
  __shared__ float sbuf[256];
  float m = -3.4e38f;
  for (size_t idx = (size_t)blockIdx.x * 256 + threadIdx.x; idx < (size_t)M_ * H_;
       idx += (size_t)gridDim.x * 256)
    m = fmaxf(m, ntv[idx & (H_ - 1)] * q[idx]);
  m = block_maxr(m, sbuf);
  if (threadIdx.x == 0) part[blockIdx.x] = m;
}

__global__ __launch_bounds__(256)
void max_final(const float* __restrict__ part, float* __restrict__ outmax)
{
  __shared__ float sbuf[256];
  int t = threadIdx.x;
  float m = fmaxf(fmaxf(part[t], part[t + 256]), fmaxf(part[t + 512], part[t + 768]));
  m = block_maxr(m, sbuf);
  if (t == 0) outmax[0] = m;
}

// per-row: h = o*(ctv*q/max) + lskip; LN; *= right; LN; -> bf16
__global__ __launch_bounds__(256)
void fuse_ht(const float* __restrict__ og, const float* __restrict__ qm,
             const float* __restrict__ ctv, const float* __restrict__ maxp,
             const float* __restrict__ lsk, const float* __restrict__ rt,
             unsigned short* __restrict__ outb)
{
  __shared__ float sbuf[256];
  int row = blockIdx.x, t = threadIdx.x;
  float inv = 1.f / maxp[0];
  float vals[4];
  float s = 0.f, s2 = 0.f;
  for (int i = 0; i < 4; ++i) {
    int col = t + (i << 8);
    size_t idx = (size_t)row * H_ + col;
    float h = og[idx] * (ctv[col] * qm[idx] * inv) + lsk[idx];
    vals[i] = h; s += h; s2 += h * h;
  }
  s  = block_sum(s,  sbuf);
  s2 = block_sum(s2, sbuf);
  float mean = s / H_;
  float rstd = rsqrtf(s2 / H_ - mean * mean + EPSL);
  float s3 = 0.f, s4 = 0.f;
  for (int i = 0; i < 4; ++i) {
    size_t idx = (size_t)row * H_ + t + (i << 8);
    float v = (vals[i] - mean) * rstd * rt[idx];
    vals[i] = v; s3 += v; s4 += v * v;
  }
  s3 = block_sum(s3, sbuf);
  s4 = block_sum(s4, sbuf);
  float mean2 = s3 / H_;
  float rstd2 = rsqrtf(s4 / H_ - mean2 * mean2 + EPSL);
  for (int i = 0; i < 4; ++i)
    outb[(size_t)row * H_ + t + (i << 8)] = f2bf((vals[i] - mean2) * rstd2);
}

// ---------------- host-side orchestration ----------------
extern "C" void kernel_launch(void* const* d_in, const int* in_sizes, int n_in,
                              void* d_out, int out_size, void* d_ws, size_t ws_size,
                              hipStream_t stream) {
  const float* x      = (const float*)d_in[0];
  const float* W_left = (const float*)d_in[1];
  const float* b_left = (const float*)d_in[2];
  const float* W_right= (const float*)d_in[3];
  const float* b_right= (const float*)d_in[4];
  const float* conv_w = (const float*)d_in[5];
  const float* conv_b = (const float*)d_in[6];
  const float* W_skip = (const float*)d_in[7];
  const float* b_skip = (const float*)d_in[8];
  const float* Wq     = (const float*)d_in[9];
  const float* bq     = (const float*)d_in[10];
  const float* Wk     = (const float*)d_in[11];
  const float* bk     = (const float*)d_in[12];
  const float* Wv     = (const float*)d_in[13];
  const float* bv     = (const float*)d_in[14];
  const float* W_i    = (const float*)d_in[15];
  const float* b_i    = (const float*)d_in[16];
  const float* W_f    = (const float*)d_in[17];
  const float* b_f    = (const float*)d_in[18];
  const float* W_o    = (const float*)d_in[19];
  const float* b_o    = (const float*)d_in[20];
  const float* W_proj = (const float*)d_in[21];
  const float* b_proj = (const float*)d_in[22];
  float* out = (float*)d_out;

  char* ws = (char*)d_ws;
  size_t cur = 0;
  auto alloc = [&](size_t bytes) -> void* {
    void* p = ws + cur;
    cur = (cur + bytes + 255) & ~(size_t)255;
    return p;
  };

  // bf16 weights
  unsigned short* wl_bf   = (unsigned short*)alloc((size_t)H_ * E_ * 2);
  unsigned short* wr_bf   = (unsigned short*)alloc((size_t)H_ * E_ * 2);
  unsigned short* wsk_bf  = (unsigned short*)alloc((size_t)H_ * H_ * 2);
  unsigned short* wi_bf   = (unsigned short*)alloc((size_t)H_ * H_ * 2);
  unsigned short* wf_bf   = (unsigned short*)alloc((size_t)H_ * H_ * 2);
  unsigned short* wo_bf   = (unsigned short*)alloc((size_t)H_ * H_ * 2);
  unsigned short* wp_bf   = (unsigned short*)alloc((size_t)E_ * H_ * 2);
  unsigned short* wq_bf   = (unsigned short*)alloc((size_t)D_ * BLK_ * BLK_ * 2);
  unsigned short* wk_bf   = (unsigned short*)alloc((size_t)D_ * BLK_ * BLK_ * 2);
  unsigned short* wv_bf   = (unsigned short*)alloc((size_t)D_ * BLK_ * BLK_ * 2);
  unsigned short* cw_bf   = (unsigned short*)alloc((size_t)KW * H_ * H_ * 2);
  // activations
  unsigned short* xn_bf   = (unsigned short*)alloc((size_t)M_ * E_ * 2);
  float*          left_f  = (float*)alloc((size_t)M_ * H_ * 4);
  unsigned short* left_bf = (unsigned short*)alloc((size_t)M_ * H_ * 2);
  float*          right_f = (float*)alloc((size_t)M_ * H_ * 4);
  float*          ll_f    = (float*)alloc((size_t)M_ * H_ * 4);
  unsigned short* ll_bf   = (unsigned short*)alloc((size_t)M_ * H_ * 2);
  float*          lskip_f = (float*)alloc((size_t)M_ * H_ * 4);
  float*          q_f     = (float*)alloc((size_t)M_ * H_ * 4);
  float*          k_f     = (float*)alloc((size_t)M_ * H_ * 4);
  float*          v_f     = (float*)alloc((size_t)M_ * H_ * 4);
  float*          ig_f    = (float*)alloc((size_t)M_ * H_ * 4);
  float*          fg_f    = (float*)alloc((size_t)M_ * H_ * 4);
  float*          og_f    = (float*)alloc((size_t)M_ * H_ * 4);
  float*          ctnt_v  = (float*)alloc((size_t)2 * H_ * 4);   // ct then nt
  float*          ctv     = ctnt_v;
  float*          ntv     = ctnt_v + H_;
  float*          maxpart = (float*)alloc(1024 * 4);
  float*          maxval  = (float*)alloc(256);
  unsigned short* fused_bf= (unsigned short*)alloc((size_t)M_ * H_ * 2);
  float*          proj_f  = (float*)alloc((size_t)M_ * E_ * 4);

  // 1) weight casts
  cast_f32_bf16<<<1024, 256, 0, stream>>>(W_left, wl_bf, (size_t)H_ * E_);
  cast_f32_bf16<<<1024, 256, 0, stream>>>(W_right, wr_bf, (size_t)H_ * E_);
  cast_f32_bf16<<<1024, 256, 0, stream>>>(W_skip, wsk_bf, (size_t)H_ * H_);
  cast_f32_bf16<<<1024, 256, 0, stream>>>(W_i, wi_bf, (size_t)H_ * H_);
  cast_f32_bf16<<<1024, 256, 0, stream>>>(W_f, wf_bf, (size_t)H_ * H_);
  cast_f32_bf16<<<1024, 256, 0, stream>>>(W_o, wo_bf, (size_t)H_ * H_);
  cast_f32_bf16<<<1024, 256, 0, stream>>>(W_proj, wp_bf, (size_t)E_ * H_);
  cast_f32_bf16<<<256, 256, 0, stream>>>(Wq, wq_bf, (size_t)D_ * BLK_ * BLK_);
  cast_f32_bf16<<<256, 256, 0, stream>>>(Wk, wk_bf, (size_t)D_ * BLK_ * BLK_);
  cast_f32_bf16<<<256, 256, 0, stream>>>(Wv, wv_bf, (size_t)D_ * BLK_ * BLK_);
  conv_transpose_cast<<<8192, 256, 0, stream>>>(conv_w, cw_bf);

  // 2) xn = LN(x) -> bf16
  ln_rows<<<M_, 256, 0, stream>>>(x, nullptr, xn_bf, E_);

  // 3) left / right GEMMs  (M=2048, N=1024, K=512)
  gemm_bf16_wmma<<<dim3(H_ / 128, M_ / 128), 256, 0, stream>>>(
      xn_bf, E_, wl_bf, E_, b_left, left_f, left_bf, H_, E_, 0);
  gemm_bf16_wmma<<<dim3(H_ / 128, M_ / 128), 256, 0, stream>>>(
      xn_bf, E_, wr_bf, E_, b_right, right_f, nullptr, H_, E_, 1);

  // 4) causal conv (dominant GEMM chain), fused silu -> left_left
  conv_causal_wmma<<<dim3(H_ / 128, M_ / 128), 256, 0, stream>>>(
      left_bf, cw_bf, conv_b, ll_f, ll_bf);

  // 5) l_skip
  gemm_bf16_wmma<<<dim3(H_ / 128, M_ / 128), 256, 0, stream>>>(
      ll_bf, H_, wsk_bf, H_, b_skip, lskip_f, nullptr, H_, H_, 0);

  // 6) block-diagonal q, k (from left_left), v (from left)
  for (int d = 0; d < D_; ++d) {
    int co = d * BLK_;
    gemm_bf16_wmma<<<dim3(BLK_ / 128, M_ / 128), 256, 0, stream>>>(
        ll_bf + co, H_, wq_bf + (size_t)d * BLK_ * BLK_, BLK_, bq + co,
        q_f + co, nullptr, H_, BLK_, 0);
    gemm_bf16_wmma<<<dim3(BLK_ / 128, M_ / 128), 256, 0, stream>>>(
        ll_bf + co, H_, wk_bf + (size_t)d * BLK_ * BLK_, BLK_, bk + co,
        k_f + co, nullptr, H_, BLK_, 0);
    gemm_bf16_wmma<<<dim3(BLK_ / 128, M_ / 128), 256, 0, stream>>>(
        left_bf + co, H_, wv_bf + (size_t)d * BLK_ * BLK_, BLK_, bv + co,
        v_f + co, nullptr, H_, BLK_, 0);
  }

  // 7) gates: GEMM then LN+exp / LN+sigmoid
  gemm_bf16_wmma<<<dim3(H_ / 128, M_ / 128), 256, 0, stream>>>(
      ll_bf, H_, wi_bf, H_, b_i, ig_f, nullptr, H_, H_, 0);
  gemm_bf16_wmma<<<dim3(H_ / 128, M_ / 128), 256, 0, stream>>>(
      ll_bf, H_, wf_bf, H_, b_f, fg_f, nullptr, H_, H_, 0);
  gemm_bf16_wmma<<<dim3(H_ / 128, M_ / 128), 256, 0, stream>>>(
      ll_bf, H_, wo_bf, H_, b_o, og_f, nullptr, H_, H_, 0);
  ln_act<<<M_, 256, 0, stream>>>(ig_f, 1);
  ln_act<<<M_, 256, 0, stream>>>(fg_f, 1);   // kept for fidelity (multiplies zero state)
  ln_act<<<M_, 256, 0, stream>>>(og_f, 2);

  // 8) ct / nt column means of row-LN'd products
  zero_f32<<<8, 256, 0, stream>>>(ctnt_v, 2 * H_);
  ctnt_accum<<<M_, 256, 0, stream>>>(ig_f, v_f, k_f, ctv, ntv);

  // 9) global max of nt*q
  max_partial<<<1024, 256, 0, stream>>>(q_f, ntv, maxpart);
  max_final<<<1, 256, 0, stream>>>(maxpart, maxval);

  // 10) ht + skip -> LN -> * right -> LN -> bf16
  fuse_ht<<<M_, 256, 0, stream>>>(og_f, q_f, ctv, maxval, lskip_f, right_f, fused_bf);

  // 11) projection GEMM (M=2048, N=512, K=1024) then final LN into d_out
  gemm_bf16_wmma<<<dim3(E_ / 128, M_ / 128), 256, 0, stream>>>(
      fused_bf, H_, wp_bf, H_, b_proj, proj_f, nullptr, E_, H_, 0);
  ln_rows<<<M_, 256, 0, stream>>>(proj_f, out, nullptr, E_);
}
